// DAHead_11192684773464
// MI455X (gfx1250) — matrix operand order repaired and checked by reading, not compiled, not run on hardware
//
#include <hip/hip_runtime.h>
#include <hip/hip_bf16.h>

// ---------------------------------------------------------------------------
// DANet head (PAM + CAM) for MI455X / gfx1250, wave32, WMMA bf16->f32.
// All heavy contractions go through v_wmma_f32_16x16x32_bf16.
// Matrix kernels: 64x128 block tile, 4 waves of 32x64 (2x4 fragments),
// double-buffered LDS filled by GLOBAL_LOAD_ASYNC_TO_LDS_B128 (ASYNCcnt),
// so no register staging -> fits the backend's 128-VGPR ceiling, no spills.
// One s_wait_asynccnt + one barrier per K-step; async fills for the next
// buffer overlap the WMMAs on the current one.
// ---------------------------------------------------------------------------

typedef __bf16 bf16;
typedef __attribute__((ext_vector_type(16))) __bf16 v16bf;
typedef __attribute__((ext_vector_type(8)))  __bf16 v8bf;
typedef __attribute__((ext_vector_type(8)))  float  v8f;

#define KPAD 40   // LDS row pitch in bf16 (80B): 16-lane row-strided frag reads hit 16 banks
#define MT 64
#define NT 128
#define MTHREADS 128   // threads per block for matrix kernels (4 waves)

__device__ inline v8bf zero_v8bf() {
    v8bf z;
#pragma unroll
    for (int i = 0; i < 8; ++i) z[i] = (bf16)0.0f;
    return z;
}

// Async DMA 16B global -> LDS (no VGPR data path, ASYNCcnt-tracked).
__device__ inline void async_ld_b128(unsigned lds_off, const void* gaddr) {
    asm volatile("global_load_async_to_lds_b128 %0, %1, off"
                 :: "v"(lds_off), "v"(gaddr) : "memory");
}
__device__ inline void wait_async0() {
    asm volatile("s_wait_asynccnt 0x0" ::: "memory");
}

// ---------------------------------------------------------------------------
// Layout converters / weight packers
// ---------------------------------------------------------------------------

// NCHW f32 -> NHWC bf16 (per batch [C][4096] -> [4096][C]), tiled transpose.
__global__ void k_nchw2nhwc_bf16(const float* __restrict__ in, bf16* __restrict__ out, int C) {
    __shared__ float t[32][33];
    const int b = blockIdx.z;
    const int p0 = blockIdx.x * 32, c0 = blockIdx.y * 32;
    const int tx = threadIdx.x, ty = threadIdx.y;  // block (32,8)
#pragma unroll
    for (int j = 0; j < 32; j += 8)
        t[ty + j][tx] = in[((size_t)b * C + (c0 + ty + j)) * 4096 + p0 + tx];
    __syncthreads();
#pragma unroll
    for (int j = 0; j < 32; j += 8)
        out[((size_t)b * 4096 + (p0 + ty + j)) * C + c0 + tx] = (bf16)t[tx][ty + j];
}

// bf16 [rows][cols] -> bf16 [cols][rows], tiled transpose.
__global__ void k_transpose_bf16(const bf16* __restrict__ in, bf16* __restrict__ out,
                                 int rows, int cols) {
    __shared__ bf16 t[32][33];
    const int r0 = blockIdx.x * 32, c0 = blockIdx.y * 32;
    const int tx = threadIdx.x, ty = threadIdx.y;  // block (32,8)
#pragma unroll
    for (int j = 0; j < 32; j += 8)
        t[ty + j][tx] = in[(size_t)(r0 + ty + j) * cols + c0 + tx];
    __syncthreads();
#pragma unroll
    for (int j = 0; j < 32; j += 8)
        out[(size_t)(c0 + ty + j) * rows + r0 + tx] = t[tx][ty + j];
}

// OIHW f32 [Co][Ci][3][3] -> bf16 [tap(9)][Co][Ci]  (K contiguous per output row)
__global__ void k_pack_w3x3(const float* __restrict__ in, bf16* __restrict__ out,
                            int Co, int Ci) {
    const size_t total = (size_t)9 * Co * Ci;
    size_t i = (size_t)blockIdx.x * blockDim.x + threadIdx.x;
    if (i >= total) return;
    const int ci = (int)(i % Ci);
    const int co = (int)((i / Ci) % Co);
    const int tap = (int)(i / ((size_t)Ci * Co));
    out[i] = (bf16)in[((size_t)co * Ci + ci) * 9 + tap];
}

__global__ void k_cast_bf16(const float* __restrict__ in, bf16* __restrict__ out, size_t n) {
    size_t i = (size_t)blockIdx.x * blockDim.x + threadIdx.x;
    if (i < n) out[i] = (bf16)in[i];
}

// ---------------------------------------------------------------------------
// 3x3 conv as implicit GEMM with WMMA. Activations NHWC bf16, weights packed
// [tap][Co][Ci] bf16.  Fused scale/bias/ReLU epilogue, NHWC bf16 output.
// Double-buffered LDS, async-DMA fills, one wait+barrier per K-step.
// ---------------------------------------------------------------------------
__global__ __launch_bounds__(MTHREADS)
void k_conv3x3_wmma(
    const bf16* __restrict__ xT,   // [B*4096][Ci] NHWC
    const bf16* __restrict__ wpk,  // [9][Co][Ci]
    const float* __restrict__ scale, const float* __restrict__ bias,
    bf16* __restrict__ out,        // [B*4096][Co] NHWC
    int Ci, int Co) {
    __shared__ bf16 As[2][MT * KPAD];
    __shared__ bf16 Bs[2][NT * KPAD];

    const int tid  = threadIdx.x;
    const int lane = tid & 31, wid = tid >> 5;        // 4 waves
    const int mW = (wid & 1) * 32, nW = (wid >> 1) * 64;  // wave tile 32x64
    const int khalf = lane >> 4, lm = lane & 15;
    const int n0 = blockIdx.x * NT;    // pixel tile (global, batch-major)
    const int m0 = blockIdx.y * MT;    // out-channel tile

    v8f acc[2][4];
#pragma unroll
    for (int mi = 0; mi < 2; ++mi)
#pragma unroll
        for (int ni = 0; ni < 4; ++ni)
#pragma unroll
            for (int i = 0; i < 8; ++i) acc[mi][ni][i] = 0.0f;

    // A fill: 64 rows x 32 ch, thread -> (row tid/2, 16-bf16 half tid%2)
    const int rA = tid >> 1, segA = tid & 1;
    // B fill: 128 pixels x 32 ch, thread -> full 32-bf16 row
    const int pIdx = n0 + tid;
    const int pb = pIdx >> 12, ph = (pIdx >> 6) & 63, pw = pIdx & 63;

    const int stepsPerTap = Ci >> 5;
    const int nSteps = 9 * stepsPerTap;

    // Issue async DMA fills of LDS buffer `buf` for K-step s.
    auto issueStep = [&](int s, int buf) {
        const int tap = s / stepsPerTap;                 // uniform scalar
        const int kc  = (s - tap * stepsPerTap) << 5;
        const int dh  = tap / 3 - 1;
        const int dw  = tap - (tap / 3) * 3 - 1;
        const char* ap = (const char*)(wpk + (size_t)(tap * Co + m0 + rA) * Ci + kc) + segA * 32;
        const unsigned la = (unsigned)(uintptr_t)&As[buf][rA * KPAD + segA * 16];
        async_ld_b128(la, ap);
        async_ld_b128(la + 16, ap + 16);
        const int hs = ph + dh, ws = pw + dw;
        const unsigned lb = (unsigned)(uintptr_t)&Bs[buf][tid * KPAD];
        if (hs >= 0 && hs < 64 && ws >= 0 && ws < 64) {
            const char* p = (const char*)(xT + (size_t)((pb << 12) + (hs << 6) + ws) * Ci + kc);
            async_ld_b128(lb, p);
            async_ld_b128(lb + 16, p + 16);
            async_ld_b128(lb + 32, p + 32);
            async_ld_b128(lb + 48, p + 48);
        } else {
            v8bf z = zero_v8bf();
            bf16* d = &Bs[buf][tid * KPAD];
            ((v8bf*)d)[0] = z; ((v8bf*)d)[1] = z;
            ((v8bf*)d)[2] = z; ((v8bf*)d)[3] = z;
        }
    };

    issueStep(0, 0);
    int cur = 0;
    for (int s = 0; s < nSteps; ++s) {
        wait_async0();        // my async fills of buf[cur] landed in LDS
        __syncthreads();      // everyone's fills landed (also drains zero ds_stores)
        if (s + 1 < nSteps) issueStep(s + 1, cur ^ 1);  // overlaps WMMAs below
        // A fragments (16x32): lane<16 K {0..7,16..23}, lane>=16 K {8..15,24..31}
        v16bf af[2];
#pragma unroll
        for (int mi = 0; mi < 2; ++mi) {
            const bf16* arow = &As[cur][(mW + mi * 16 + lm) * KPAD];
            ((v8bf*)&af[mi])[0] = *(const v8bf*)(arow + 8 * khalf);
            ((v8bf*)&af[mi])[1] = *(const v8bf*)(arow + 16 + 8 * khalf);
        }
#pragma unroll
        for (int ni = 0; ni < 4; ++ni) {
            const bf16* brow = &Bs[cur][(nW + ni * 16 + lm) * KPAD];
            v16bf b;
            ((v8bf*)&b)[0] = *(const v8bf*)(brow + 8 * khalf);
            ((v8bf*)&b)[1] = *(const v8bf*)(brow + 16 + 8 * khalf);
#pragma unroll
            for (int mi = 0; mi < 2; ++mi)
                acc[mi][ni] = __builtin_amdgcn_wmma_f32_16x16x32_bf16(
                    false, af[mi], false, b, (short)0, acc[mi][ni], false, false);
        }
        cur ^= 1;
    }
    // Epilogue: D lane layout — VGPR r, lane l: M = r + 8*(l>>4), N = l&15
#pragma unroll
    for (int mi = 0; mi < 2; ++mi)
#pragma unroll
        for (int ni = 0; ni < 4; ++ni) {
            const int n = n0 + nW + ni * 16 + lm;
#pragma unroll
            for (int r = 0; r < 8; ++r) {
                const int m = m0 + mW + mi * 16 + r + 8 * khalf;
                float v = acc[mi][ni][r] * scale[m] + bias[m];
                v = v > 0.0f ? v : 0.0f;
                out[(size_t)n * Co + m] = (bf16)v;
            }
        }
}

// ---------------------------------------------------------------------------
// Generic batched WMMA GEMM:  C[m,n] = sum_k A[m*lda+k] * B[n*ldb+k]
// mode 0: f32 out[m*ldo+n] (+bias)         -- classifier heads
// mode 1: bf16 out[m*ldo+n] (+bias)        -- fd / attention logits
// mode 2: bf16 out[n*ldo+m] (+bias,*scale,+res[n*ldo+m]) -- NHWC + residual
// Same 64x128 / 32x64-wave async-DMA pipelined scheme as the conv.
// ---------------------------------------------------------------------------
__global__ __launch_bounds__(MTHREADS)
void k_gemm_wmma(
    const bf16* __restrict__ A, const bf16* __restrict__ B,
    int M, int N, int K, int lda, int ldb,
    long long sA, long long sB,
    float* __restrict__ outF, bf16* __restrict__ outB,
    int mode, int ldo, long long sO,
    const float* __restrict__ bias, const float* __restrict__ scalePtr,
    const bf16* __restrict__ res, long long sR) {
    A += (size_t)blockIdx.z * sA;
    B += (size_t)blockIdx.z * sB;

    __shared__ bf16 As[2][MT * KPAD];
    __shared__ bf16 Bs[2][NT * KPAD];

    const int tid  = threadIdx.x;
    const int lane = tid & 31, wid = tid >> 5;
    const int mW = (wid & 1) * 32, nW = (wid >> 1) * 64;
    const int khalf = lane >> 4, lm = lane & 15;
    const int n0 = blockIdx.x * NT;
    const int m0 = blockIdx.y * MT;

    v8f acc[2][4];
#pragma unroll
    for (int mi = 0; mi < 2; ++mi)
#pragma unroll
        for (int ni = 0; ni < 4; ++ni)
#pragma unroll
            for (int i = 0; i < 8; ++i) acc[mi][ni][i] = 0.0f;

    const int rA = tid >> 1, segA = tid & 1;
    const int rowA = m0 + rA;
    const int colB = n0 + tid;
    const bool aOK = rowA < M, bOK = colB < N;

    auto issueStep = [&](int s, int buf) {
        const int kc = s << 5;
        const unsigned la = (unsigned)(uintptr_t)&As[buf][rA * KPAD + segA * 16];
        if (aOK) {
            const char* ap = (const char*)(A + (size_t)rowA * lda + kc) + segA * 32;
            async_ld_b128(la, ap);
            async_ld_b128(la + 16, ap + 16);
        } else {
            v8bf z = zero_v8bf();
            bf16* a = &As[buf][rA * KPAD + segA * 16];
            ((v8bf*)a)[0] = z; ((v8bf*)a)[1] = z;
        }
        const unsigned lb = (unsigned)(uintptr_t)&Bs[buf][tid * KPAD];
        if (bOK) {
            const char* p = (const char*)(B + (size_t)colB * ldb + kc);
            async_ld_b128(lb, p);
            async_ld_b128(lb + 16, p + 16);
            async_ld_b128(lb + 32, p + 32);
            async_ld_b128(lb + 48, p + 48);
        } else {
            v8bf z = zero_v8bf();
            bf16* d = &Bs[buf][tid * KPAD];
            ((v8bf*)d)[0] = z; ((v8bf*)d)[1] = z;
            ((v8bf*)d)[2] = z; ((v8bf*)d)[3] = z;
        }
    };

    const int nSteps = K >> 5;
    issueStep(0, 0);
    int cur = 0;
    for (int s = 0; s < nSteps; ++s) {
        wait_async0();
        __syncthreads();
        if (s + 1 < nSteps) issueStep(s + 1, cur ^ 1);
        v16bf af[2];
#pragma unroll
        for (int mi = 0; mi < 2; ++mi) {
            const bf16* arow = &As[cur][(mW + mi * 16 + lm) * KPAD];
            ((v8bf*)&af[mi])[0] = *(const v8bf*)(arow + 8 * khalf);
            ((v8bf*)&af[mi])[1] = *(const v8bf*)(arow + 16 + 8 * khalf);
        }
#pragma unroll
        for (int ni = 0; ni < 4; ++ni) {
            const bf16* brow = &Bs[cur][(nW + ni * 16 + lm) * KPAD];
            v16bf b;
            ((v8bf*)&b)[0] = *(const v8bf*)(brow + 8 * khalf);
            ((v8bf*)&b)[1] = *(const v8bf*)(brow + 16 + 8 * khalf);
#pragma unroll
            for (int mi = 0; mi < 2; ++mi)
                acc[mi][ni] = __builtin_amdgcn_wmma_f32_16x16x32_bf16(
                    false, af[mi], false, b, (short)0, acc[mi][ni], false, false);
        }
        cur ^= 1;
    }

    const size_t ob = (size_t)blockIdx.z * sO;
#pragma unroll
    for (int mi = 0; mi < 2; ++mi)
#pragma unroll
        for (int ni = 0; ni < 4; ++ni) {
            const int n = n0 + nW + ni * 16 + lm;
            if (n >= N) continue;
#pragma unroll
            for (int r = 0; r < 8; ++r) {
                const int m = m0 + mW + mi * 16 + r + 8 * khalf;
                if (m >= M) continue;
                float v = acc[mi][ni][r];
                if (bias) v += bias[m];
                if (scalePtr) v *= *scalePtr;
                if (mode == 2) {
                    const size_t idx = (size_t)n * ldo + m;
                    if (res) v += (float)res[(size_t)blockIdx.z * sR + idx];
                    outB[ob + idx] = (bf16)v;
                } else if (mode == 1) {
                    outB[ob + (size_t)m * ldo + n] = (bf16)v;
                } else {
                    outF[ob + (size_t)m * ldo + n] = v;
                }
            }
        }
}

// ---------------------------------------------------------------------------
// In-place row softmax over bf16 rows.
// mode 0: softmax(a);  mode 1: softmax(rowmax - a) == exp(rowmin - a)/sum
// ---------------------------------------------------------------------------
__global__ __launch_bounds__(256) void k_softmax_bf16(bf16* __restrict__ data,
                                                      int len, int mode) {
    __shared__ float red[256];
    const size_t base = (size_t)blockIdx.x * len;
    const int tid = threadIdx.x;
    float r = mode ? 3.402823466e+38f : -3.402823466e+38f;
    for (int i = tid; i < len; i += 256) {
        float v = (float)data[base + i];
        r = mode ? fminf(r, v) : fmaxf(r, v);
    }
    red[tid] = r;
    __syncthreads();
    for (int s = 128; s > 0; s >>= 1) {
        if (tid < s) red[tid] = mode ? fminf(red[tid], red[tid + s])
                                     : fmaxf(red[tid], red[tid + s]);
        __syncthreads();
    }
    const float m = red[0];
    __syncthreads();
    float sum = 0.0f;
    for (int i = tid; i < len; i += 256) {
        float v = (float)data[base + i];
        sum += __expf(mode ? (m - v) : (v - m));
    }
    red[tid] = sum;
    __syncthreads();
    for (int s = 128; s > 0; s >>= 1) {
        if (tid < s) red[tid] += red[tid + s];
        __syncthreads();
    }
    const float inv = 1.0f / red[0];
    for (int i = tid; i < len; i += 256) {
        float v = (float)data[base + i];
        data[base + i] = (bf16)(__expf(mode ? (m - v) : (v - m)) * inv);
    }
}

__global__ void k_add_bf16(const bf16* __restrict__ a, const bf16* __restrict__ b,
                           bf16* __restrict__ o, size_t n) {
    size_t i = (size_t)blockIdx.x * blockDim.x + threadIdx.x;
    if (i < n) o[i] = (bf16)((float)a[i] + (float)b[i]);
}

// Bilinear 2x upsample (half-pixel).  in layout: [c][b*4096 + h*64 + w] f32.
// out layout: [b][C][128][128] f32.
__global__ void k_upsample2x(const float* __restrict__ in, float* __restrict__ out,
                             size_t total, int C) {
    size_t t = (size_t)blockIdx.x * blockDim.x + threadIdx.x;
    if (t >= total) return;
    const int ow = (int)(t & 127);
    const int oh = (int)((t >> 7) & 127);
    const int c  = (int)((t >> 14) % C);
    const int b  = (int)(t / ((size_t)C << 14));
    const float sy = (oh + 0.5f) * 0.5f - 0.5f;
    const float sx = (ow + 0.5f) * 0.5f - 0.5f;
    const int y0 = (int)floorf(sy), x0 = (int)floorf(sx);
    const float fy = sy - y0, fx = sx - x0;
    const int y0c = min(max(y0, 0), 63), y1c = min(max(y0 + 1, 0), 63);
    const int x0c = min(max(x0, 0), 63), x1c = min(max(x0 + 1, 0), 63);
    const float* base = in + (size_t)c * 16384 + (size_t)b * 4096;
    const float v00 = base[y0c * 64 + x0c], v01 = base[y0c * 64 + x1c];
    const float v10 = base[y1c * 64 + x0c], v11 = base[y1c * 64 + x1c];
    out[t] = (1.0f - fy) * ((1.0f - fx) * v00 + fx * v01) +
             fy * ((1.0f - fx) * v10 + fx * v11);
}

// ---------------------------------------------------------------------------
// Host orchestration
// ---------------------------------------------------------------------------
extern "C" void kernel_launch(void* const* d_in, const int* in_sizes, int n_in,
                              void* d_out, int out_size, void* d_ws, size_t ws_size,
                              hipStream_t stream) {
    (void)in_sizes; (void)n_in; (void)out_size; (void)ws_size;
    const int B = 4, Cin = 2048, IC = 512, C8 = 64, NC = 81;
    const int HW = 4096, Npix = B * HW;  // 16384

    const float* x    = (const float*)d_in[0];
    const float* wp1  = (const float*)d_in[1];
    const float* sp1  = (const float*)d_in[2];
    const float* bp1  = (const float*)d_in[3];
    const float* wc1  = (const float*)d_in[4];
    const float* sc1  = (const float*)d_in[5];
    const float* bc1  = (const float*)d_in[6];
    const float* pwb  = (const float*)d_in[7];
    const float* pbb  = (const float*)d_in[8];
    const float* pwc  = (const float*)d_in[9];
    const float* pbc  = (const float*)d_in[10];
    const float* pwd  = (const float*)d_in[11];
    const float* pbd  = (const float*)d_in[12];
    const float* alp  = (const float*)d_in[13];
    const float* bet  = (const float*)d_in[14];
    const float* wp2  = (const float*)d_in[15];
    const float* sp2  = (const float*)d_in[16];
    const float* bp2  = (const float*)d_in[17];
    const float* wc2  = (const float*)d_in[18];
    const float* sc2  = (const float*)d_in[19];
    const float* bc2  = (const float*)d_in[20];
    const float* wlog = (const float*)d_in[21];
    const float* blog = (const float*)d_in[22];
    const float* wp3  = (const float*)d_in[23];
    const float* bp3  = (const float*)d_in[24];
    const float* wc3  = (const float*)d_in[25];
    const float* bc3  = (const float*)d_in[26];
    float* out = (float*)d_out;

    size_t off = 0;
    auto alloc = [&](size_t bytes) -> void* {
        void* p = (char*)d_ws + off;
        off += (bytes + 255) & ~(size_t)255;
        return p;
    };
    bf16* xT     = (bf16*)alloc((size_t)Npix * Cin * 2);
    bf16* wpkP   = (bf16*)alloc((size_t)9 * IC * Cin * 2);
    bf16* wpkC   = (bf16*)alloc((size_t)9 * IC * Cin * 2);
    bf16* featP  = (bf16*)alloc((size_t)Npix * IC * 2);
    bf16* featC  = (bf16*)alloc((size_t)Npix * IC * 2);
    bf16* wbB    = (bf16*)alloc((size_t)C8 * IC * 2);
    bf16* wcB    = (bf16*)alloc((size_t)C8 * IC * 2);
    bf16* wdB    = (bf16*)alloc((size_t)IC * IC * 2);
    bf16* wlogB  = (bf16*)alloc((size_t)NC * IC * 2);
    bf16* wp3B   = (bf16*)alloc((size_t)NC * IC * 2);
    bf16* wc3B   = (bf16*)alloc((size_t)NC * IC * 2);
    bf16* fb     = (bf16*)alloc((size_t)Npix * C8 * 2);
    bf16* fc     = (bf16*)alloc((size_t)Npix * C8 * 2);
    bf16* fd     = (bf16*)alloc((size_t)IC * Npix * 2);
    bf16* attn   = (bf16*)alloc((size_t)B * HW * HW * 2);
    bf16* pamO   = (bf16*)alloc((size_t)Npix * IC * 2);
    bf16* wpkP2  = (bf16*)alloc((size_t)9 * IC * IC * 2);
    bf16* wpkC2  = (bf16*)alloc((size_t)9 * IC * IC * 2);
    bf16* featP2 = (bf16*)alloc((size_t)Npix * IC * 2);
    bf16* fcC    = (bf16*)alloc((size_t)IC * Npix * 2);
    bf16* attC   = (bf16*)alloc((size_t)B * IC * IC * 2);
    bf16* camO   = (bf16*)alloc((size_t)Npix * IC * 2);
    bf16* featC2 = (bf16*)alloc((size_t)Npix * IC * 2);
    bf16* fusion = (bf16*)alloc((size_t)Npix * IC * 2);
    float* outP  = (float*)alloc((size_t)NC * Npix * 4);
    float* outC  = (float*)alloc((size_t)NC * Npix * 4);
    float* outFu = (float*)alloc((size_t)NC * Npix * 4);

    auto gemm = [&](const bf16* A, const bf16* Bm, int M, int N, int K, int lda, int ldb,
                    long long sA, long long sB, int batches,
                    float* oF, bf16* oB, int mode, int ldo, long long sO,
                    const float* bias, const float* sc, const bf16* res, long long sR) {
        dim3 g((N + NT - 1) / NT, (M + MT - 1) / MT, batches);
        k_gemm_wmma<<<g, MTHREADS, 0, stream>>>(A, Bm, M, N, K, lda, ldb, sA, sB,
                                                oF, oB, mode, ldo, sO, bias, sc, res, sR);
    };

    // ---- layout conversion / weight packing ----
    k_nchw2nhwc_bf16<<<dim3(4096 / 32, Cin / 32, B), dim3(32, 8), 0, stream>>>(x, xT, Cin);
    {
        size_t n1 = (size_t)9 * IC * Cin;
        k_pack_w3x3<<<(int)((n1 + 255) / 256), 256, 0, stream>>>(wp1, wpkP, IC, Cin);
        k_pack_w3x3<<<(int)((n1 + 255) / 256), 256, 0, stream>>>(wc1, wpkC, IC, Cin);
        size_t n2 = (size_t)9 * IC * IC;
        k_pack_w3x3<<<(int)((n2 + 255) / 256), 256, 0, stream>>>(wp2, wpkP2, IC, IC);
        k_pack_w3x3<<<(int)((n2 + 255) / 256), 256, 0, stream>>>(wc2, wpkC2, IC, IC);
    }
    k_cast_bf16<<<(C8 * IC + 255) / 256, 256, 0, stream>>>(pwb, wbB, (size_t)C8 * IC);
    k_cast_bf16<<<(C8 * IC + 255) / 256, 256, 0, stream>>>(pwc, wcB, (size_t)C8 * IC);
    k_cast_bf16<<<(IC * IC + 255) / 256, 256, 0, stream>>>(pwd, wdB, (size_t)IC * IC);
    k_cast_bf16<<<(NC * IC + 255) / 256, 256, 0, stream>>>(wlog, wlogB, (size_t)NC * IC);
    k_cast_bf16<<<(NC * IC + 255) / 256, 256, 0, stream>>>(wp3, wp3B, (size_t)NC * IC);
    k_cast_bf16<<<(NC * IC + 255) / 256, 256, 0, stream>>>(wc3, wc3B, (size_t)NC * IC);

    // ---- stage 1: two 3x3 convs 2048->512 with BN+ReLU ----
    k_conv3x3_wmma<<<dim3(Npix / NT, IC / MT), MTHREADS, 0, stream>>>(xT, wpkP, sp1, bp1, featP, Cin, IC);
    k_conv3x3_wmma<<<dim3(Npix / NT, IC / MT), MTHREADS, 0, stream>>>(xT, wpkC, sc1, bc1, featC, Cin, IC);

    // ---- PAM ----
    gemm(wbB, featP, C8, Npix, IC, IC, IC, 0, 0, 1,
         nullptr, fb, 2, C8, 0, pbb, nullptr, nullptr, 0);                       // fb [n][64]
    gemm(wcB, featP, C8, Npix, IC, IC, IC, 0, 0, 1,
         nullptr, fc, 2, C8, 0, pbc, nullptr, nullptr, 0);                       // fc [m][64]
    gemm(wdB, featP, IC, Npix, IC, IC, IC, 0, 0, 1,
         nullptr, fd, 1, Npix, 0, pbd, nullptr, nullptr, 0);                     // fd [c][bHW]
    gemm(fb, fc, HW, HW, C8, C8, C8, (long long)HW * C8, (long long)HW * C8, B,
         nullptr, attn, 1, HW, (long long)HW * HW, nullptr, nullptr, nullptr, 0); // logits[n][m]
    k_softmax_bf16<<<B * HW, 256, 0, stream>>>(attn, HW, 0);
    gemm(fd, attn, IC, HW, HW, Npix, HW, (long long)HW, (long long)HW * HW, B,
         nullptr, pamO, 2, IC, (long long)HW * IC, nullptr, alp, featP,
         (long long)HW * IC);                                                    // alpha*fe+featP
    k_conv3x3_wmma<<<dim3(Npix / NT, IC / MT), MTHREADS, 0, stream>>>(pamO, wpkP2, sp2, bp2, featP2, IC, IC);

    // ---- CAM ----
    k_transpose_bf16<<<dim3(Npix / 32, IC / 32), dim3(32, 8), 0, stream>>>(featC, fcC, Npix, IC);
    gemm(fcC, fcC, IC, IC, HW, Npix, Npix, (long long)HW, (long long)HW, B,
         nullptr, attC, 1, IC, (long long)IC * IC, nullptr, nullptr, nullptr, 0); // att[c][d]
    k_softmax_bf16<<<B * IC, 256, 0, stream>>>(attC, IC, 1);
    gemm(attC, featC, IC, HW, IC, IC, IC, (long long)IC * IC, (long long)HW * IC, B,
         nullptr, camO, 2, IC, (long long)HW * IC, nullptr, bet, featC,
         (long long)HW * IC);                                                    // beta*fe+featC
    k_conv3x3_wmma<<<dim3(Npix / NT, IC / MT), MTHREADS, 0, stream>>>(camO, wpkC2, sc2, bc2, featC2, IC, IC);

    // ---- fusion + classifier heads (M=81, bounds-guarded) ----
    k_add_bf16<<<(int)(((size_t)Npix * IC + 255) / 256), 256, 0, stream>>>(
        featP2, featC2, fusion, (size_t)Npix * IC);
    gemm(wp3B, featP2, NC, Npix, IC, IC, IC, 0, 0, 1,
         outP, nullptr, 0, Npix, 0, bp3, nullptr, nullptr, 0);
    gemm(wc3B, featC2, NC, Npix, IC, IC, IC, 0, 0, 1,
         outC, nullptr, 0, Npix, 0, bc3, nullptr, nullptr, 0);
    gemm(wlogB, fusion, NC, Npix, IC, IC, IC, 0, 0, 1,
         outFu, nullptr, 0, Npix, 0, blog, nullptr, nullptr, 0);

    // ---- 2x bilinear upsample into d_out (p, c, fusion) ----
    const size_t upN = (size_t)B * NC * 128 * 128;  // 5,308,416
    k_upsample2x<<<(int)((upN + 255) / 256), 256, 0, stream>>>(outP, out, upN, NC);
    k_upsample2x<<<(int)((upN + 255) / 256), 256, 0, stream>>>(outC, out + upN, upN, NC);
    k_upsample2x<<<(int)((upN + 255) / 256), 256, 0, stream>>>(outFu, out + 2 * upN, upN, NC);
}
